// ClusterMemory_14611478741436
// MI455X (gfx1250) — matrix operand order repaired and checked by reading, not compiled, not run on hardware
//
#include <hip/hip_runtime.h>
#include <hip/hip_bf16.h>

// Problem constants (from reference)
#define BB 512
#define NN 65536
#define DD 256
#define INV_TEMP 20.0f   // 1/0.05; also the static logit upper bound (unit vectors)

typedef __attribute__((ext_vector_type(16))) __bf16        v16bf;
typedef __attribute__((ext_vector_type(8)))  float         v8f;
typedef __attribute__((ext_vector_type(8)))  unsigned short v8us;
typedef __attribute__((ext_vector_type(16))) unsigned short v16us;
typedef __attribute__((ext_vector_type(4)))  int           v4i;

// CDNA5 async global->LDS path (ASYNCcnt-tracked). Guarded so both toolchains
// and the host pass stay compile-clean; fallback is the proven sync fill.
// Builtin signature (from hipcc diagnostic): (v4i addrspace(1)*, v4i
// addrspace(3)*, imm int offset, imm int cpol).
#if __has_builtin(__builtin_amdgcn_global_load_async_to_lds_b128) && \
    __has_builtin(__builtin_amdgcn_s_wait_asynccnt)
#define USE_ASYNC 1
typedef __attribute__((address_space(1))) v4i gv4i;
typedef __attribute__((address_space(3))) v4i lv4i;
#define ASYNC_CP16(gp, lp, off)                                          \
  __builtin_amdgcn_global_load_async_to_lds_b128(                        \
      (gv4i*)(gp), (lv4i*)(lp), (off), 0)
#else
#define USE_ASYNC 0
#endif

__device__ __forceinline__ unsigned short bf16_rne(float f) {
  unsigned u = __float_as_uint(f);
  u += 0x7FFFu + ((u >> 16) & 1u);
  return (unsigned short)(u >> 16);
}
__device__ __forceinline__ float bf16_up(unsigned short h) {
  return __uint_as_float(((unsigned)h) << 16);
}

// ---------------------------------------------------------------------------
// Kernel 1: per-row normalize of inputs, split into bf16 hi/lo, compute the
// target logit 20*<x_b, f_{t_b}>, and zero the per-row sumexp accumulator.
// One block per batch row (256 threads == D).
// ---------------------------------------------------------------------------
__global__ void __launch_bounds__(256)
prep_kernel(const float* __restrict__ inputs,
            const float* __restrict__ features,
            const int*   __restrict__ targets,
            unsigned short* __restrict__ xhi,
            unsigned short* __restrict__ xlo,
            float* __restrict__ row_sum,
            float* __restrict__ row_tgt) {
  __shared__ float red[256];
  const int b = blockIdx.x;
  const int t = threadIdx.x;

  float v = inputs[b * DD + t];
  red[t] = v * v;
  __syncthreads();
  for (int s = 128; s > 0; s >>= 1) {
    if (t < s) red[t] += red[t + s];
    __syncthreads();
  }
  const float rn = rsqrtf(red[0]);
  __syncthreads();

  const float x = v * rn;
  const unsigned short h = bf16_rne(x);
  const unsigned short l = bf16_rne(x - bf16_up(h));
  xhi[b * DD + t] = h;
  xlo[b * DD + t] = l;

  const int tgt = targets[b];
  const float fv = features[(long)tgt * DD + t];
  red[t] = x * fv;
  __syncthreads();
  for (int s = 128; s > 0; s >>= 1) {
    if (t < s) red[t] += red[t + s];
    __syncthreads();
  }
  if (t == 0) {
    row_tgt[b] = INV_TEMP * red[0];
    row_sum[b] = 0.0f;
  }
}

// ---------------------------------------------------------------------------
// Kernel 2: main fused GEMM + exp + per-row partial reduction.
// Block = 256 threads = 8 waves. Block tile: 128 rows (B) x 128 cols (N).
// Wave w owns rows [r0 + 16w, r0 + 16w + 16) and all 128 cols (8 WMMA tiles).
// K loop: 32-wide slabs of features. The raw fp32 slab is double-buffered in
// LDS via GLOBAL_LOAD_ASYNC_TO_LDS_B128 (issue slab k+1, s_wait_asynccnt for
// slab k), hiding global latency behind the WMMA stage. Each thread
// async-copies exactly the 64B it later converts, so copy->convert needs no
// barrier. 3 WMMAs/tile = bf16 hi/lo fp32 emulation.
// ---------------------------------------------------------------------------
__global__ void __launch_bounds__(256)
logits_kernel(const unsigned short* __restrict__ xhi,
              const unsigned short* __restrict__ xlo,
              const float* __restrict__ features,
              float* __restrict__ row_sum) {
  __shared__ __align__(16) unsigned short lds_hi[128 * 32];
  __shared__ __align__(16) unsigned short lds_lo[128 * 32];
#if USE_ASYNC
  __shared__ __align__(16) float slab[2][128 * 32];
#endif

  const int tid   = threadIdx.x;
  const int wave  = tid >> 5;
  const int lane  = tid & 31;
  const int lhalf = lane >> 4;   // 0 or 1
  const int l15   = lane & 15;
  const int n0 = blockIdx.x * 128;
  const int r0 = blockIdx.y * 128 + wave * 16;

  v8f acc[8];
#pragma unroll
  for (int t = 0; t < 8; ++t)
#pragma unroll
    for (int i = 0; i < 8; ++i) acc[t][i] = 0.0f;

  // A-fragment base pointers (k order used here, same as B: lanes 0-15 hold K{0..15},
  // lanes 16-31 hold K{16..31} of row M = lane&15).
  const unsigned short* xh = xhi + (long)(r0 + l15) * DD;
  const unsigned short* xl = xlo + (long)(r0 + l15) * DD;

  // This thread's private 16-float region of each feature slab.
  const int col = tid >> 1;
  const int kh  = (tid & 1) * 16;
  const float* gsrc = features + (long)(n0 + col) * DD + kh;  // + k0 at use

#if USE_ASYNC
  {  // prologue: slab 0 <- k0 = 0
    float* lp = &slab[0][col * 32 + kh];
    ASYNC_CP16(gsrc, lp, 0);
    ASYNC_CP16(gsrc, lp, 16);
    ASYNC_CP16(gsrc, lp, 32);
    ASYNC_CP16(gsrc, lp, 48);
  }
#endif

  for (int kt = 0; kt < DD / 32; ++kt) {
    const int k0 = kt * 32;
    float4 f0, f1, f2, f3;
#if USE_ASYNC
    if (kt < DD / 32 - 1) {  // issue next slab before waiting on current one
      const float* gp = gsrc + k0 + 32;
      float* lp = &slab[(kt + 1) & 1][col * 32 + kh];
      ASYNC_CP16(gp, lp, 0);
      ASYNC_CP16(gp, lp, 16);
      ASYNC_CP16(gp, lp, 32);
      ASYNC_CP16(gp, lp, 48);
      __builtin_amdgcn_s_wait_asynccnt(4);  // current slab's 4 copies landed
    } else {
      __builtin_amdgcn_s_wait_asynccnt(0);
    }
    __syncthreads();  // all waves done reading bf16 slab from previous iter
    const float* sp = &slab[kt & 1][col * 32 + kh];
    f0 = ((const float4*)sp)[0];
    f1 = ((const float4*)sp)[1];
    f2 = ((const float4*)sp)[2];
    f3 = ((const float4*)sp)[3];
#else
    __syncthreads();  // all waves done reading bf16 slab from previous iter
    const float4* fp = (const float4*)(gsrc + k0);
    f0 = fp[0]; f1 = fp[1]; f2 = fp[2]; f3 = fp[3];
#endif
    {
      // Convert this thread's 16 floats -> bf16 hi/lo, store to shared slab.
      float fv[16];
      fv[0]=f0.x; fv[1]=f0.y; fv[2]=f0.z; fv[3]=f0.w;
      fv[4]=f1.x; fv[5]=f1.y; fv[6]=f1.z; fv[7]=f1.w;
      fv[8]=f2.x; fv[9]=f2.y; fv[10]=f2.z; fv[11]=f2.w;
      fv[12]=f3.x; fv[13]=f3.y; fv[14]=f3.z; fv[15]=f3.w;
      v8us hv0, hv1, lv0, lv1;
#pragma unroll
      for (int j = 0; j < 8; ++j) {
        unsigned short h0 = bf16_rne(fv[j]);
        hv0[j] = h0;
        lv0[j] = bf16_rne(fv[j] - bf16_up(h0));
        unsigned short h1 = bf16_rne(fv[8 + j]);
        hv1[j] = h1;
        lv1[j] = bf16_rne(fv[8 + j] - bf16_up(h1));
      }
      const int base = col * 32 + kh;
      *(v8us*)&lds_hi[base]     = hv0;
      *(v8us*)&lds_hi[base + 8] = hv1;
      *(v8us*)&lds_lo[base]     = lv0;
      *(v8us*)&lds_lo[base + 8] = lv1;
    }
    __syncthreads();

    // A fragments (direct from global; 0.5 MB total, stays L2-hot)
    const int kA = k0 + lhalf * 16;
    v8us ah0 = *(const v8us*)(xh + kA);
    v8us ah1 = *(const v8us*)(xh + kA + 8);
    v8us al0 = *(const v8us*)(xl + kA);
    v8us al1 = *(const v8us*)(xl + kA + 8);
    v16us Ah, Al;
#pragma unroll
    for (int i = 0; i < 8; ++i) {
      Ah[i] = ah0[i]; Ah[8 + i] = ah1[i];
      Al[i] = al0[i]; Al[8 + i] = al1[i];
    }
    const v16bf Ahb = __builtin_bit_cast(v16bf, Ah);
    const v16bf Alb = __builtin_bit_cast(v16bf, Al);

#pragma unroll
    for (int t = 0; t < 8; ++t) {
      // B k order used here (same as A): lanes 0-15 = col (lane&15), K 0..15;
      // lanes 16-31 = same cols, K 16..31; contiguous K within a lane.
      const int bbase = (t * 16 + l15) * 32 + lhalf * 16;
      v8us bh0 = *(const v8us*)&lds_hi[bbase];
      v8us bh1 = *(const v8us*)&lds_hi[bbase + 8];
      v8us bl0 = *(const v8us*)&lds_lo[bbase];
      v8us bl1 = *(const v8us*)&lds_lo[bbase + 8];
      v16us Bh, Bl;
#pragma unroll
      for (int i = 0; i < 8; ++i) {
        Bh[i] = bh0[i]; Bh[8 + i] = bh1[i];
        Bl[i] = bl0[i]; Bl[8 + i] = bl1[i];
      }
      const v16bf Bhb = __builtin_bit_cast(v16bf, Bh);
      const v16bf Blb = __builtin_bit_cast(v16bf, Bl);

      acc[t] = __builtin_amdgcn_wmma_f32_16x16x32_bf16(
          false, Ahb, false, Bhb, (short)0, acc[t], false, false);
      acc[t] = __builtin_amdgcn_wmma_f32_16x16x32_bf16(
          false, Ahb, false, Blb, (short)0, acc[t], false, false);
      acc[t] = __builtin_amdgcn_wmma_f32_16x16x32_bf16(
          false, Alb, false, Bhb, (short)0, acc[t], false, false);
    }
  }

  // Epilogue: exp with the static max bound (logits <= 20 exactly), then
  // per-row partial sums. C/D layout: VGPR i, lanes 0-15 -> row i, lanes
  // 16-31 -> row i+8; lane&15 = column.
  float part[8];
#pragma unroll
  for (int i = 0; i < 8; ++i) part[i] = 0.0f;
#pragma unroll
  for (int t = 0; t < 8; ++t)
#pragma unroll
    for (int i = 0; i < 8; ++i)
      part[i] += __expf(INV_TEMP * acc[t][i] - INV_TEMP);

#pragma unroll
  for (int off = 1; off < 16; off <<= 1)
#pragma unroll
    for (int i = 0; i < 8; ++i)
      part[i] += __shfl_xor(part[i], off, 32);

  if (l15 == 0) {
    const int rbase = r0 + lhalf * 8;
#pragma unroll
    for (int i = 0; i < 8; ++i)
      atomicAdd(&row_sum[rbase + i], part[i]);
  }
}

// ---------------------------------------------------------------------------
// Kernel 3: loss = mean_b( log(sumexp_b) + 20 - tgt_logit_b )
// ---------------------------------------------------------------------------
__global__ void __launch_bounds__(256)
finalize_kernel(const float* __restrict__ row_sum,
                const float* __restrict__ row_tgt,
                float* __restrict__ out) {
  __shared__ float red[256];
  const int t = threadIdx.x;
  float s = 0.0f;
  for (int b = t; b < BB; b += 256)
    s += (logf(row_sum[b]) + INV_TEMP) - row_tgt[b];
  red[t] = s;
  __syncthreads();
  for (int st = 128; st > 0; st >>= 1) {
    if (t < st) red[t] += red[t + st];
    __syncthreads();
  }
  if (t == 0) out[0] = red[0] / (float)BB;
}

extern "C" void kernel_launch(void* const* d_in, const int* in_sizes, int n_in,
                              void* d_out, int out_size, void* d_ws, size_t ws_size,
                              hipStream_t stream) {
  (void)in_sizes; (void)n_in; (void)out_size; (void)ws_size;
  const float* inputs   = (const float*)d_in[0];
  // d_in[1] = ema_inputs: unused by the loss (reference never reads it)
  const int*   targets  = (const int*)d_in[2];
  const float* features = (const float*)d_in[3];

  unsigned short* xhi = (unsigned short*)d_ws;           // B*D bf16
  unsigned short* xlo = xhi + (size_t)BB * DD;           // B*D bf16
  float* row_sum = (float*)(xlo + (size_t)BB * DD);      // B floats
  float* row_tgt = row_sum + BB;                         // B floats

  prep_kernel<<<BB, 256, 0, stream>>>(inputs, features, targets,
                                      xhi, xlo, row_sum, row_tgt);

  dim3 grid(NN / 128, BB / 128);
  logits_kernel<<<grid, 256, 0, stream>>>(xhi, xlo, features, row_sum);

  finalize_kernel<<<1, 256, 0, stream>>>(row_sum, row_tgt, (float*)d_out);
}
